// RGAT_DGL_27822798143912
// MI455X (gfx1250) — matrix operand order repaired and checked by reading, not compiled, not run on hardware
//
#include <hip/hip_runtime.h>
#include <stdint.h>

// ---------------------------------------------------------------------------
// Types for CDNA5 WMMA (gfx1250, wave32)
// ---------------------------------------------------------------------------
typedef __attribute__((ext_vector_type(16))) __bf16 v16bf;
typedef __attribute__((ext_vector_type(8)))  float  v8f;

union FragBF16 {
    int4  q[2];   // 32 bytes = 16 bf16
    v16bf v;
};

#define N_NODES 40000
#define N_EDGES 150000
#define NHEADS  4
#define HDIM    128
#define FEATS   512   // NHEADS*HDIM

// ---------------------------------------------------------------------------
// Utility kernels
// ---------------------------------------------------------------------------
__global__ void k_fill_f32(float* p, float v, long n) {
    long i = (long)blockIdx.x * blockDim.x + threadIdx.x;
    if (i < n) p[i] = v;
}

__device__ __forceinline__ unsigned short f32_to_bf16(float f) {
    unsigned u = __float_as_uint(f);
    u += 0x7FFFu + ((u >> 16) & 1u);       // round-to-nearest-even
    return (unsigned short)(u >> 16);
}

__global__ void k_cvt_f32_bf16(const float* __restrict__ in,
                               unsigned short* __restrict__ out, long n) {
    long i = (long)blockIdx.x * blockDim.x + threadIdx.x;
    if (i < n) out[i] = f32_to_bf16(in[i]);
}

// Transpose W[K x N] (f32, row-major) -> Wt[N x K] (bf16, row-major)
__global__ void k_transpose_bf16(const float* __restrict__ W,
                                 unsigned short* __restrict__ Wt,
                                 int K, int N) {
    long i = (long)blockIdx.x * blockDim.x + threadIdx.x;
    long total = (long)K * N;
    if (i >= total) return;
    int n = (int)(i % N);
    int k = (int)(i / N);
    Wt[(size_t)n * K + k] = f32_to_bf16(W[(size_t)k * N + n]);
}

// ---------------------------------------------------------------------------
// WMMA GEMM: C[M x N](f32) = A[M x K](bf16 row-major) * Bt[N x K]^T (bf16)
// One wave -> 32(M) x 64(N) output tile (2 A-frags x 4 B-frags = 8 WMMAs per
// K-step of 32); block = 8 waves sharing one 32-row band of A (L2/WGP$ hits).
// A fragment (16x32 bf16) per ISA 7.12.2: lane l (l15=l&15,lh=l>>4) holds
// row M=l15, K = {k0+8*lh .. +7} in elems 0..7 and {k0+16+8*lh ..} in 8..15.
// B fragment: lane l = column l15; same K order as A (elems 0..7 = k0+8*lh.., 8..15 = k0+16+8*lh..).
// C/D: VGPR r -> M = r + 8*lh, N = l15.
// ---------------------------------------------------------------------------
__global__ __launch_bounds__(256) void k_gemm_wmma_bf16(
        const unsigned short* __restrict__ A,
        const unsigned short* __restrict__ Bt,
        float* __restrict__ C,
        int M, int N, int K,
        const float* __restrict__ bias) {
    const int lane  = threadIdx.x & 31;
    const int wave  = threadIdx.x >> 5;
    const int l15   = lane & 15;
    const int lh    = lane >> 4;

    const int mTile  = blockIdx.y;            // 32 rows per tile
    const int nStrip = blockIdx.x * 8 + wave; // 64 cols per strip
    const int row0 = mTile * 32;
    const int col0 = nStrip * 64;
    if (col0 >= N) return;

    const int arow0 = row0 + l15;
    const int arow1 = row0 + 16 + l15;
    const unsigned short* Arow0 = A + (size_t)(arow0 < M ? arow0 : 0) * K;
    const unsigned short* Arow1 = A + (size_t)(arow1 < M ? arow1 : 0) * K;

    const unsigned short* Brow[4];
#pragma unroll
    for (int t = 0; t < 4; ++t) {
        int col = col0 + t * 16 + l15;
        Brow[t] = Bt + (size_t)(col < N ? col : 0) * K;
    }

    v8f acc0[4] = {};
    v8f acc1[4] = {};

#pragma unroll 2
    for (int k0 = 0; k0 < K; k0 += 32) {
        const int kA = k0 + lh * 8;
        const int kB = k0 + lh * 8;

        FragBF16 a0, a1;
        a0.q[0] = *(const int4*)(Arow0 + kA);
        a0.q[1] = *(const int4*)(Arow0 + kA + 16);
        a1.q[0] = *(const int4*)(Arow1 + kA);
        a1.q[1] = *(const int4*)(Arow1 + kA + 16);

        FragBF16 b[4];
#pragma unroll
        for (int t = 0; t < 4; ++t) {
            b[t].q[0] = *(const int4*)(Brow[t] + kB);
            b[t].q[1] = *(const int4*)(Brow[t] + kB + 16);
        }

#pragma unroll
        for (int t = 0; t < 4; ++t) {
            acc0[t] = __builtin_amdgcn_wmma_f32_16x16x32_bf16(
                          false, a0.v, false, b[t].v,
                          (short)0, acc0[t], false, false);
            acc1[t] = __builtin_amdgcn_wmma_f32_16x16x32_bf16(
                          false, a1.v, false, b[t].v,
                          (short)0, acc1[t], false, false);
        }
    }

#pragma unroll
    for (int t = 0; t < 4; ++t) {
        int col = col0 + t * 16 + l15;
        if (col >= N) continue;
        float badd = bias ? bias[col] : 0.0f;
#pragma unroll
        for (int r = 0; r < 8; ++r) {
            int row = row0 + lh * 8 + r;
            if (row < M) C[(size_t)row * N + col] = acc0[t][r] + badd;
            int row2 = row + 16;
            if (row2 < M) C[(size_t)row2 * N + col] = acc1[t][r] + badd;
        }
    }
}

// ---------------------------------------------------------------------------
// GAT edge kernels
// ---------------------------------------------------------------------------
// el[n,h] = <feat[n,h,:], al[h,:]>, er likewise. One thread per (n,h).
__global__ void k_el_er(const float* __restrict__ feat,
                        const float* __restrict__ al,
                        const float* __restrict__ ar,
                        float* __restrict__ el, float* __restrict__ er) {
    long i = (long)blockIdx.x * blockDim.x + threadIdx.x;
    if (i >= (long)N_NODES * NHEADS) return;
    int h = (int)(i & 3);
    const float4* f = (const float4*)(feat + (size_t)i * HDIM);
    const float4* a = (const float4*)(al + h * HDIM);
    const float4* b = (const float4*)(ar + h * HDIM);
    float sl = 0.f, sr = 0.f;
#pragma unroll 4
    for (int j = 0; j < HDIM / 4; ++j) {
        float4 fv = f[j], av = a[j], bv = b[j];
        sl += fv.x*av.x + fv.y*av.y + fv.z*av.z + fv.w*av.w;
        sr += fv.x*bv.x + fv.y*bv.y + fv.z*bv.z + fv.w*bv.w;
    }
    el[i] = sl; er[i] = sr;
}

__device__ __forceinline__ unsigned f32_ord_enc(float f) {
    unsigned u = __float_as_uint(f);
    return (u & 0x80000000u) ? ~u : (u | 0x80000000u);
}
__device__ __forceinline__ float f32_ord_dec(unsigned u) {
    unsigned b = (u & 0x80000000u) ? (u & 0x7FFFFFFFu) : ~u;
    return __uint_as_float(b);
}

// e = leaky_relu(el[src]+er[dst], 0.2); segment-max into mu[dst] (ordered u32).
__global__ void k_edge_logits(const int* __restrict__ src,
                              const int* __restrict__ dst,
                              const float* __restrict__ el,
                              const float* __restrict__ er,
                              float* __restrict__ ebuf,
                              unsigned* __restrict__ mu) {
    long i = (long)blockIdx.x * blockDim.x + threadIdx.x;
    if (i >= (long)N_EDGES * NHEADS) return;
    int e = (int)(i >> 2), h = (int)(i & 3);
    int s = src[e], d = dst[e];
    float v = el[(size_t)s * NHEADS + h] + er[(size_t)d * NHEADS + h];
    v = (v > 0.f) ? v : 0.2f * v;
    ebuf[i] = v;
    atomicMax(&mu[(size_t)d * NHEADS + h], f32_ord_enc(v));
}

// w = exp(e - m[dst]); segment-sum into sbuf[dst]
__global__ void k_edge_expsum(const int* __restrict__ dst,
                              float* __restrict__ ebuf,
                              const unsigned* __restrict__ mu,
                              float* __restrict__ sbuf) {
    long i = (long)blockIdx.x * blockDim.x + threadIdx.x;
    if (i >= (long)N_EDGES * NHEADS) return;
    int e = (int)(i >> 2), h = (int)(i & 3);
    int d = dst[e];
    float m = f32_ord_dec(mu[(size_t)d * NHEADS + h]);
    float w = __expf(ebuf[i] - m);
    ebuf[i] = w;
    atomicAdd(&sbuf[(size_t)d * NHEADS + h], w);
}

// agg[dst] += feat[src] * (w / s[dst]); one block (128 thr) per edge, float4/thr
__global__ __launch_bounds__(128) void k_aggregate(
        const int* __restrict__ src, const int* __restrict__ dst,
        const float* __restrict__ feat, const float* __restrict__ ebuf,
        const float* __restrict__ sbuf, float* __restrict__ agg) {
    int e = blockIdx.x;
    int s = src[e], d = dst[e];
    int d4 = threadIdx.x;          // 0..127 -> 4 floats each
    int h  = d4 >> 5;              // 32 float4 per head (128 floats)
    float sum = sbuf[(size_t)d * NHEADS + h];
    float a = ebuf[(size_t)e * NHEADS + h] / sum;
    float4 fv = ((const float4*)(feat + (size_t)s * FEATS))[d4];
    float* out = agg + (size_t)d * FEATS + d4 * 4;
    atomicAdd(out + 0, fv.x * a);
    atomicAdd(out + 1, fv.y * a);
    atomicAdd(out + 2, fv.z * a);
    atomicAdd(out + 3, fv.w * a);
}

// accum += act(agg + b)  (per-type bias + optional leaky_relu(0.01) pre-sum)
__global__ void k_bias_act_accum(const float* __restrict__ agg,
                                 const float* __restrict__ b,
                                 float* __restrict__ accum, int activate) {
    long i = (long)blockIdx.x * blockDim.x + threadIdx.x;
    if (i >= (long)N_NODES * FEATS) return;
    float v = agg[i] + b[i & (FEATS - 1)];
    if (activate) v = (v > 0.f) ? v : 0.01f * v;
    accum[i] += v;
}

// ---------------------------------------------------------------------------
// Host orchestration
// ---------------------------------------------------------------------------
static inline size_t align256(size_t x) { return (x + 255) & ~(size_t)255; }

extern "C" void kernel_launch(void* const* d_in, const int* in_sizes, int n_in,
                              void* d_out, int out_size, void* d_ws, size_t ws_size,
                              hipStream_t stream) {
    (void)in_sizes; (void)n_in; (void)out_size; (void)ws_size;

    const float* x    = (const float*)d_in[0];
    const int* srcs[2] = { (const int*)d_in[1], (const int*)d_in[3] };
    const int* dsts[2] = { (const int*)d_in[2], (const int*)d_in[4] };
    const float* Wl[3]  = { (const float*)d_in[5],  (const float*)d_in[9],  (const float*)d_in[13] };
    const float* all[3] = { (const float*)d_in[6],  (const float*)d_in[10], (const float*)d_in[14] };
    const float* arl[3] = { (const float*)d_in[7],  (const float*)d_in[11], (const float*)d_in[15] };
    const float* bl[3]  = { (const float*)d_in[8],  (const float*)d_in[12], (const float*)d_in[16] };
    const float* Wout = (const float*)d_in[17];
    const float* bout = (const float*)d_in[18];

    const int NCLS = 2983;

    // workspace layout
    uint8_t* ws = (uint8_t*)d_ws;
    size_t off = 0;
    unsigned short* in_bf16 = (unsigned short*)(ws + off); off += align256((size_t)N_NODES * 1024 * 2);
    unsigned short* wt_bf16 = (unsigned short*)(ws + off); off += align256((size_t)NCLS * FEATS * 2);
    float* feat  = (float*)(ws + off); off += align256((size_t)N_NODES * FEATS * 4);
    float* agg   = (float*)(ws + off); off += align256((size_t)N_NODES * FEATS * 4);
    float* accum = (float*)(ws + off); off += align256((size_t)N_NODES * FEATS * 4);
    float* el    = (float*)(ws + off); off += align256((size_t)N_NODES * NHEADS * 4);
    float* er    = (float*)(ws + off); off += align256((size_t)N_NODES * NHEADS * 4);
    unsigned* mu = (unsigned*)(ws + off); off += align256((size_t)N_NODES * NHEADS * 4);
    float* sbuf  = (float*)(ws + off); off += align256((size_t)N_NODES * NHEADS * 4);
    float* ebuf  = (float*)(ws + off); off += align256((size_t)N_EDGES * NHEADS * 4);

    const int TPB = 256;
    auto blocks = [](long n, int t) { return (int)((n + t - 1) / t); };

    // x (f32) -> bf16
    {
        long n = (long)N_NODES * 1024;
        k_cvt_f32_bf16<<<blocks(n, TPB), TPB, 0, stream>>>(x, in_bf16, n);
    }

    const int mTiles = (N_NODES + 31) / 32;   // 1250

    for (int l = 0; l < 3; ++l) {
        const int K = (l == 0) ? 1024 : FEATS;
        const int activate = (l < 2) ? 1 : 0;

        // accum = 0
        {
            long n = (long)N_NODES * FEATS;
            k_fill_f32<<<blocks(n, TPB), TPB, 0, stream>>>(accum, 0.0f, n);
        }

        for (int t = 0; t < 2; ++t) {
            // transpose+convert weights: W[K x 512] -> Wt[512 x K] bf16
            {
                long n = (long)K * FEATS;
                k_transpose_bf16<<<blocks(n, TPB), TPB, 0, stream>>>(
                    Wl[l] + (size_t)t * K * FEATS, wt_bf16, K, FEATS);
            }
            // feat = in_bf16 @ W   (M=40000, N=512, K)
            {
                dim3 grid(1 /* 512/64/8 */, mTiles);
                k_gemm_wmma_bf16<<<grid, 256, 0, stream>>>(
                    in_bf16, wt_bf16, feat, N_NODES, FEATS, K, nullptr);
            }
            // attention coefficients
            {
                long n = (long)N_NODES * NHEADS;
                k_el_er<<<blocks(n, TPB), TPB, 0, stream>>>(
                    feat, all[l] + (size_t)t * FEATS, arl[l] + (size_t)t * FEATS, el, er);
                k_fill_f32<<<blocks(n, TPB), TPB, 0, stream>>>((float*)mu, 0.0f, n);
                k_fill_f32<<<blocks(n, TPB), TPB, 0, stream>>>(sbuf, 0.0f, n);
            }
            {
                long n = (long)N_NODES * FEATS;
                k_fill_f32<<<blocks(n, TPB), TPB, 0, stream>>>(agg, 0.0f, n);
            }
            {
                long ne = (long)N_EDGES * NHEADS;
                k_edge_logits<<<blocks(ne, TPB), TPB, 0, stream>>>(
                    srcs[t], dsts[t], el, er, ebuf, mu);
                k_edge_expsum<<<blocks(ne, TPB), TPB, 0, stream>>>(
                    dsts[t], ebuf, mu, sbuf);
            }
            k_aggregate<<<N_EDGES, 128, 0, stream>>>(
                srcs[t], dsts[t], feat, ebuf, sbuf, agg);
            {
                long n = (long)N_NODES * FEATS;
                k_bias_act_accum<<<blocks(n, TPB), TPB, 0, stream>>>(
                    agg, bl[l] + (size_t)t * FEATS, accum, activate);
            }
        }

        // next-layer input (bf16) <- accum
        {
            long n = (long)N_NODES * FEATS;
            k_cvt_f32_bf16<<<blocks(n, TPB), TPB, 0, stream>>>(accum, in_bf16, n);
        }
    }

    // classifier: d_out = accum_bf16 @ Wout + bout   (M=40000, N=2983, K=512)
    {
        long n = (long)FEATS * NCLS;
        k_transpose_bf16<<<blocks(n, TPB), TPB, 0, stream>>>(Wout, wt_bf16, FEATS, NCLS);
        int nStrips = (NCLS + 63) / 64;               // 47
        dim3 grid((nStrips + 7) / 8, mTiles);         // 6 x 1250
        k_gemm_wmma_bf16<<<grid, 256, 0, stream>>>(
            in_bf16, wt_bf16, (float*)d_out, N_NODES, NCLS, FEATS, bout);
    }
}
